// ContextualModule_74887049773166
// MI455X (gfx1250) — matrix operand order repaired and checked by reading, not compiled, not run on hardware
//
#include <hip/hip_runtime.h>

// ---------------------------------------------------------------------------
// Contextual attention (inpainting) for MI455X / gfx1250.
// Both heavy phases are bf16 WMMA GEMMs with f32 accumulation:
//   phase A: score[4096,4096] = K[4096,576] @ FP^T  (FP stored pixel-major)
//   phase B: rec[64,4096]     = sum_s K_s^T[64,4096] @ att_shift_s[4096,4096]
// WMMA fragments are contiguous ds_load_b128 pairs; gemm1 is a register-staged
// double-buffered LDS pipeline (1 barrier/iter, HBM latency overlapped).
// ---------------------------------------------------------------------------

typedef __attribute__((ext_vector_type(16))) __bf16          v16bf;
typedef __attribute__((ext_vector_type(8)))  float           v8f;
typedef __attribute__((ext_vector_type(8)))  unsigned short  us8;

#define HW   64
#define LTOT 4096        // h*w
#define CCH  64
#define KD   576         // c*9

__device__ __forceinline__ unsigned short f2bf(float f) {
  unsigned int u = __builtin_bit_cast(unsigned int, f);
  unsigned int r = u + 0x7FFFu + ((u >> 16) & 1u);   // round-to-nearest-even
  return (unsigned short)(r >> 16);
}

struct Pair16 { us8 lo, hi; };

// Two contiguous 16B LDS reads -> one v16bf fragment (no per-half packing).
__device__ __forceinline__ v16bf load_frag2(const unsigned short* S, int o0, int o1) {
  Pair16 p;
  p.lo = *reinterpret_cast<const us8*>(S + o0);
  p.hi = *reinterpret_cast<const us8*>(S + o1);
  return __builtin_bit_cast(v16bf, p);
}

// A-matrix 16x32 bf16 fragment from row-major [m][lda] tile (ISA 7.12.2):
//   halfs 0..7  = K (half*8 .. +7), halfs 8..15 = K (16+half*8 .. +7)
__device__ __forceinline__ v16bf load_a_frag(const unsigned short* S, int lda,
                                             int mrow, int lane) {
  const int half = (lane >> 4) & 1;
  const int base = mrow * lda + (half << 3);
  return load_frag2(S, base, base + 16);
}

// B-matrix 32x16 bf16 fragment from TRANSPOSED [n][ldb] tile:
//   halfs 0..7  = K (half*8 .. +7), halfs 8..15 = K (16+half*8 .. +7)
__device__ __forceinline__ v16bf load_b_frag(const unsigned short* S, int ldb,
                                             int ncol, int lane) {
  const int kb = ((lane >> 4) & 1) << 3;
  const int base = ncol * ldb + kb;
  return load_frag2(S, base, base + 16);
}

// ---------------------------------------------------------------------------
// Prep kernels
// ---------------------------------------------------------------------------
__device__ __forceinline__ float patch_val(const float* img, int l, int s) {
  int dy = s / 3, dx = s - dy * 3;
  int y = l >> 6, x = l & 63;
  int yy = y + dy - 1, xx = x + dx - 1;
  if (yy >= 0 && yy < HW && xx >= 0 && xx < HW) return img[yy * HW + xx];
  return 0.0f;
}

// Kbf[l][c*9+s] = bg_masked patch + EPS  (A operand of GEMM1, row-major)
__global__ void prep_K_kernel(const float* __restrict__ bg,
                              const float* __restrict__ msk,
                              unsigned short* __restrict__ Kbf) {
  int e = blockIdx.x * blockDim.x + threadIdx.x;   // 0 .. 4096*576-1
  int l = e / KD, j = e - l * KD;
  int c = j / 9,  s = j - c * 9;
  int dy = s / 3, dx = s - dy * 3;
  int y = l >> 6, x = l & 63;
  int yy = y + dy - 1, xx = x + dx - 1;
  float v = 0.0f;
  if (yy >= 0 && yy < HW && xx >= 0 && xx < HW) {
    int n = yy * HW + xx;
    v = bg[c * (HW * HW) + n] * (1.0f - msk[n]);
  }
  Kbf[e] = f2bf(v + 1.1920929e-07f);
}

// KT[(s*64+c)][l] = same values re-tiled for GEMM2 A tiles (contiguous in l)
__global__ void prep_KT_kernel(const float* __restrict__ bg,
                               const float* __restrict__ msk,
                               unsigned short* __restrict__ KT) {
  int e = blockIdx.x * blockDim.x + threadIdx.x;   // 0 .. 9*64*4096-1
  int l = e & 4095;
  int sc = e >> 12;
  int c = sc & 63, s = sc >> 6;
  int dy = s / 3, dx = s - dy * 3;
  int y = l >> 6, x = l & 63;
  int yy = y + dy - 1, xx = x + dx - 1;
  float v = 0.0f;
  if (yy >= 0 && yy < HW && xx >= 0 && xx < HW) {
    int n = yy * HW + xx;
    v = bg[c * (HW * HW) + n] * (1.0f - msk[n]);
  }
  KT[e] = f2bf(v + 1.1920929e-07f);
}

// FPbf[n][c*9+s] = fg patch, PIXEL-major so GEMM1 B tiles load like A tiles
__global__ void prep_FP_kernel(const float* __restrict__ fg,
                               unsigned short* __restrict__ FPbf) {
  int e = blockIdx.x * blockDim.x + threadIdx.x;   // 0 .. 4096*576-1
  int n = e / KD, r = e - n * KD;
  int c = r / 9, s = r - c * 9;
  FPbf[e] = f2bf(patch_val(fg + c * (HW * HW), n, s));
}

__global__ void prep_mm_kernel(const float* __restrict__ msk,
                               float* __restrict__ mm) {
  int l = blockIdx.x * blockDim.x + threadIdx.x;
  int y = l >> 6, x = l & 63;
  float s = 0.0f;
  for (int dy = -1; dy <= 1; ++dy) {
    int yy = y + dy; if (yy < 0 || yy >= HW) continue;
    for (int dx = -1; dx <= 1; ++dx) {
      int xx = x + dx; if (xx < 0 || xx >= HW) continue;
      s += msk[yy * HW + xx];
    }
  }
  mm[l] = (s == 0.0f) ? 1.0f : 0.0f;
}

// ---------------------------------------------------------------------------
// GEMM1: score[4096,4096] = Kbf[4096,576] @ FPbf[4096,576]^T
// block: 256 thr (8 waves), tile 128x128, BK=32; wave -> 32x64 (2x4 wmma tiles)
// Register-staged, double-buffered LDS pipeline.
// ---------------------------------------------------------------------------
__global__ __launch_bounds__(256)
void gemm1_kernel(const unsigned short* __restrict__ A,   // [4096][576] (m,k)
                  const unsigned short* __restrict__ B,   // [4096][576] (n,k)
                  float* __restrict__ score) {
  __shared__ unsigned short As[2][128 * 40];   // [m][k] 128x32 (+8 pad), x2 buffers
  __shared__ unsigned short BsT[2][128 * 40];  // [n][k] 128x32 (+8 pad), x2 buffers
  const int tid  = threadIdx.x;
  const int w    = tid >> 5, lane = tid & 31;
  const int wm   = w & 3,    wn   = w >> 2;
  const int m0   = blockIdx.y * 128, n0 = blockIdx.x * 128;

  v8f acc[2][4] = {};

  const int row = tid >> 1, coff = (tid & 1) * 16;   // 128 rows x 32 halfs
  const unsigned short* aptr = A + (size_t)(m0 + row) * KD + coff;
  const unsigned short* bptr = B + (size_t)(n0 + row) * KD + coff;

  // prologue: stage tile k0=0 into registers
  uint4 ra0 = reinterpret_cast<const uint4*>(aptr)[0];
  uint4 ra1 = reinterpret_cast<const uint4*>(aptr)[1];
  uint4 rb0 = reinterpret_cast<const uint4*>(bptr)[0];
  uint4 rb1 = reinterpret_cast<const uint4*>(bptr)[1];

  int buf = 0;
  for (int k0 = 0; k0 < KD; k0 += 32) {
    // staged regs -> LDS[buf]
    uint4* al = reinterpret_cast<uint4*>(&As[buf][row * 40 + coff]);
    al[0] = ra0; al[1] = ra1;
    uint4* bl = reinterpret_cast<uint4*>(&BsT[buf][row * 40 + coff]);
    bl[0] = rb0; bl[1] = rb1;
    __syncthreads();

    // issue next tile's global loads (in flight during the WMMA chain)
    if (k0 + 32 < KD) {
      const uint4* ag  = reinterpret_cast<const uint4*>(aptr + k0 + 32);
      const uint4* bg2 = reinterpret_cast<const uint4*>(bptr + k0 + 32);
      ra0 = ag[0];  ra1 = ag[1];
      rb0 = bg2[0]; rb1 = bg2[1];
      __builtin_prefetch(aptr + k0 + 64, 0, 1);   // global_prefetch_b8 for k0+64
      __builtin_prefetch(bptr + k0 + 64, 0, 1);
    }

    // fragments: all loads first (distinct regs), then the 8-WMMA chain
    v16bf afr[2], bfr[4];
#pragma unroll
    for (int mt = 0; mt < 2; ++mt)
      afr[mt] = load_a_frag(As[buf], 40, wm * 32 + mt * 16 + (lane & 15), lane);
#pragma unroll
    for (int nt = 0; nt < 4; ++nt)
      bfr[nt] = load_b_frag(BsT[buf], 40, wn * 64 + nt * 16 + (lane & 15), lane);
#pragma unroll
    for (int nt = 0; nt < 4; ++nt) {
      acc[0][nt] = __builtin_amdgcn_wmma_f32_16x16x32_bf16(false, afr[0], false, bfr[nt],
                                                           (short)0, acc[0][nt], false, false);
      acc[1][nt] = __builtin_amdgcn_wmma_f32_16x16x32_bf16(false, afr[1], false, bfr[nt],
                                                           (short)0, acc[1][nt], false, false);
    }
    buf ^= 1;
  }

  const int half = lane >> 4, nn = lane & 15;
#pragma unroll
  for (int mt = 0; mt < 2; ++mt)
#pragma unroll
    for (int nt = 0; nt < 4; ++nt)
#pragma unroll
      for (int r = 0; r < 8; ++r) {
        int rrow = m0 + wm * 32 + mt * 16 + r + half * 8;
        int ccol = n0 + wn * 64 + nt * 16 + nn;
        score[(size_t)rrow * LTOT + ccol] = acc[mt][nt][r];
      }
}

// ---------------------------------------------------------------------------
// 3x3 ones smoothing over each row's 64x64 image, times mm[l]
// ---------------------------------------------------------------------------
__global__ void smooth_kernel(const float* __restrict__ score,
                              const float* __restrict__ mm,
                              float* __restrict__ s2) {
  int l = blockIdx.y;
  int n = blockIdx.x * blockDim.x + threadIdx.x;
  int y = n >> 6, x = n & 63;
  const float* src = score + (size_t)l * LTOT;
  float s = 0.0f;
  for (int dy = -1; dy <= 1; ++dy) {
    int yy = y + dy; if (yy < 0 || yy >= HW) continue;
    for (int dx = -1; dx <= 1; ++dx) {
      int xx = x + dx; if (xx < 0 || xx >= HW) continue;
      s += src[yy * HW + xx];
    }
  }
  s2[(size_t)l * LTOT + n] = s * mm[l];
}

// ---------------------------------------------------------------------------
// Column softmax stats (over L=4096 rows) + masked argmax.
// block = 64 columns x 4 row-groups.
// ---------------------------------------------------------------------------
__global__ __launch_bounds__(256)
void colreduce_kernel(const float* __restrict__ s2, const float* __restrict__ mm,
                      float* __restrict__ colmax, float* __restrict__ colsum,
                      int* __restrict__ argmx) {
  __shared__ float red[256];
  __shared__ float redv[256];
  __shared__ int   redi[256];
  const int tid = threadIdx.x;
  const int col = blockIdx.x * 64 + (tid & 63);
  const int rg  = tid >> 6;
  const float NEG = -3.402823e38f;

  float lmax = NEG;
  for (int r = rg; r < LTOT; r += 4)
    lmax = fmaxf(lmax, s2[(size_t)r * LTOT + col]);
  red[tid] = lmax;
  __syncthreads();
  if (rg == 0)
    red[tid] = fmaxf(fmaxf(red[tid], red[tid + 64]), fmaxf(red[tid + 128], red[tid + 192]));
  __syncthreads();
  const float Mx = red[tid & 63];
  __syncthreads();

  float lsum = 0.0f, bv = NEG;
  int bi = 0x7fffffff;
  for (int r = rg; r < LTOT; r += 4) {
    float v = s2[(size_t)r * LTOT + col];
    lsum += __expf(10.0f * (v - Mx));
    if (mm[r] != 0.0f) {
      if (v > bv || (v == bv && r < bi)) { bv = v; bi = r; }
    }
  }
  red[tid] = lsum; redv[tid] = bv; redi[tid] = bi;
  __syncthreads();
  if (rg == 0) {
    float S  = red[tid] + red[tid + 64] + red[tid + 128] + red[tid + 192];
    float v0 = redv[tid]; int i0 = redi[tid];
#pragma unroll
    for (int q = 1; q < 4; ++q) {
      float v1 = redv[tid + q * 64]; int i1 = redi[tid + q * 64];
      if (v1 > v0 || (v1 == v0 && i1 < i0)) { v0 = v1; i0 = i1; }
    }
    colmax[col] = Mx;
    colsum[col] = S;
    argmx[col]  = (v0 == NEG) ? 0 : i0;
  }
}

// zero the padded attention buffer ([4096][66][66] bf16, as u32 words)
__global__ void attzero_kernel(unsigned int* __restrict__ attp32) {
  size_t e = (size_t)blockIdx.x * blockDim.x + threadIdx.x;
  if (e < (size_t)LTOT * 4356 / 2) attp32[e] = 0u;
}

__global__ void attfill_kernel(const float* __restrict__ s2, const float* __restrict__ mm,
                               const float* __restrict__ colmax,
                               const float* __restrict__ colsum,
                               unsigned short* __restrict__ attp) {
  int l = blockIdx.y;
  int n = blockIdx.x * blockDim.x + threadIdx.x;
  float v = s2[(size_t)l * LTOT + n];
  float a = __expf(10.0f * (v - colmax[n])) / colsum[n] * mm[l];
  int y = n >> 6, x = n & 63;
  attp[(size_t)l * 4356 + (size_t)(y + 1) * 66 + (x + 1)] = f2bf(a);
}

// ---------------------------------------------------------------------------
// GEMM2: rec[64,4096] = sum over 9 shifts of K_s^T @ att_shift_s
// block: 256 thr (8 waves), tile 64x128; K loop = 9 shifts x 4096 (BK=32)
// A tiles from KT[(s*64+c)][l] (contiguous), B tiles transposed into LDS.
// ---------------------------------------------------------------------------
__global__ __launch_bounds__(256)
void gemm2_kernel(const unsigned short* __restrict__ KT,     // [9*64][4096]
                  const unsigned short* __restrict__ attp,   // [4096][66][66]
                  float* __restrict__ rec) {
  __shared__ unsigned short As2[64 * 40];    // [c][l-chunk]
  __shared__ unsigned short Bs2T[128 * 40];  // [n][l-chunk] (transposed)
  const int tid = threadIdx.x, w = tid >> 5, lane = tid & 31;
  const int wm = w & 3, wn = w >> 2;
  const int n0 = blockIdx.x * 128;

  v8f acc[4] = {};

  const int arow = tid >> 2, acoff = (tid & 3) * 8;   // 64 rows x 32 halfs (1 x b128)

  for (int s = 0; s < 9; ++s) {
    const int dy = s / 3, dx = s - dy * 3, sf = 8 - s;   // K[:, :, 2-dy, 2-dx]
    for (int lc = 0; lc < LTOT; lc += 32) {
      // A tile: As2[c][ll] = KT[(sf*64+c)][lc+ll]  (vectorized)
      {
        const uint4* ag = reinterpret_cast<const uint4*>(
            KT + (size_t)(sf * 64 + arow) * LTOT + lc + acoff);
        uint4* al = reinterpret_cast<uint4*>(&As2[arow * 40 + acoff]);
        al[0] = ag[0];
      }
      // B tile transposed: Bs2T[col][ll] = att_pad[lc+ll][y+dy][x+dx]
      {
        int ll = tid >> 3, cb = (tid & 7) * 16;
        size_t base = (size_t)(lc + ll) * 4356;
#pragma unroll
        for (int t = 0; t < 16; ++t) {
          int col = cb + t;
          int y = (n0 + col) >> 6, x = (n0 + col) & 63;
          Bs2T[col * 40 + ll] = attp[base + (size_t)(y + dy) * 66 + (x + dx)];
        }
      }
      __syncthreads();

      v16bf a = load_a_frag(As2, 40, wm * 16 + (lane & 15), lane);
      v16bf bfr[4];
#pragma unroll
      for (int i = 0; i < 4; ++i)
        bfr[i] = load_b_frag(Bs2T, 40, wn * 64 + i * 16 + (lane & 15), lane);
#pragma unroll
      for (int i = 0; i < 4; ++i)
        acc[i] = __builtin_amdgcn_wmma_f32_16x16x32_bf16(false, a, false, bfr[i],
                                                         (short)0, acc[i], false, false);
      __syncthreads();
    }
  }

  const int half = lane >> 4, nn = lane & 15;
#pragma unroll
  for (int i = 0; i < 4; ++i)
#pragma unroll
    for (int r = 0; r < 8; ++r) {
      int c   = wm * 16 + r + half * 8;
      int col = n0 + wn * 64 + i * 16 + nn;
      rec[(size_t)c * LTOT + col] = acc[i][r];
    }
}

// ---------------------------------------------------------------------------
// Epilogues
// ---------------------------------------------------------------------------
__global__ void final_kernel(const float* __restrict__ rec, const float* __restrict__ fg,
                             const float* __restrict__ msk, float* __restrict__ out) {
  int e = blockIdx.x * blockDim.x + threadIdx.x;   // 64*4096
  int n = e & 4095;
  float mv = msk[n];
  float r = rec[e] * mv * (1.0f / 9.0f);
  out[e] = r * mv + fg[e] * (1.0f - mv);
}

__global__ void offsets_kernel(const int* __restrict__ argmx, float* __restrict__ out2) {
  int n = blockIdx.x * blockDim.x + threadIdx.x;   // 4096
  int idx = argmx[n];
  const float times = 4096.0f / 4356.0f;           // hw / ((h+2)*(w+2))
  int off = (int)(((float)(idx + 1)) * times - 1.0f);   // truncation matches astype(int32)
  if (off < 0) off = 0;
  int oy = off >> 6, ox = off & 63;
  int y = n >> 6, x = n & 63;
  out2[n]        = (float)(oy - y);
  out2[4096 + n] = (float)(ox - x);
}

// ---------------------------------------------------------------------------
// Host launch: sequential batches, shared workspace (stream-ordered).
// ws layout (bytes):
//   0          Kbf    4,718,592
//   4718592    FPbf   4,718,592
//   9437184    KT     4,718,592
//   14155776   mm     16,384
//   14172160   argmax 16,384
//   14188544   colmax 16,384
//   14204928   colsum 16,384
//   14221312   rec    1,048,576
//   15269888   score  67,108,864   (attp aliases this region after smoothing)
//   82378752   score2 67,108,864   -> peak ~142.6 MB
// ---------------------------------------------------------------------------
extern "C" void kernel_launch(void* const* d_in, const int* in_sizes, int n_in,
                              void* d_out, int out_size, void* d_ws, size_t ws_size,
                              hipStream_t stream) {
  const float* fg  = (const float*)d_in[0];   // [4,64,64,64]
  const float* bgr = (const float*)d_in[1];   // [4,64,64,64]
  const float* msk = (const float*)d_in[2];   // [4,1,64,64]
  float* out = (float*)d_out;
  char*  ws  = (char*)d_ws;

  unsigned short* Kbf    = (unsigned short*)(ws + 0);
  unsigned short* FPbf   = (unsigned short*)(ws + 4718592);
  unsigned short* KT     = (unsigned short*)(ws + 9437184);
  float*          mm     = (float*)(ws + 14155776);
  int*            argmx  = (int*)  (ws + 14172160);
  float*          colmax = (float*)(ws + 14188544);
  float*          colsum = (float*)(ws + 14204928);
  float*          rec    = (float*)(ws + 14221312);
  float*          score  = (float*)(ws + 15269888);
  float*          score2 = (float*)(ws + 82378752);
  unsigned short* attp   = (unsigned short*)(ws + 15269888);  // reuses score (dead)

  for (int b = 0; b < 4; ++b) {
    const float* fg_b = fg  + (size_t)b * CCH * LTOT;
    const float* bg_b = bgr + (size_t)b * CCH * LTOT;
    const float* m_b  = msk + (size_t)b * LTOT;
    float* out_b  = out + (size_t)b * CCH * LTOT;
    float* out2_b = out + (size_t)4 * CCH * LTOT + (size_t)b * 2 * LTOT;

    prep_K_kernel  <<<9216, 256, 0, stream>>>(bg_b, m_b, Kbf);
    prep_KT_kernel <<<9216, 256, 0, stream>>>(bg_b, m_b, KT);
    prep_FP_kernel <<<9216, 256, 0, stream>>>(fg_b, FPbf);
    prep_mm_kernel <<<16, 256, 0, stream>>>(m_b, mm);

    gemm1_kernel   <<<dim3(32, 32), 256, 0, stream>>>(Kbf, FPbf, score);

    smooth_kernel  <<<dim3(16, 4096), 256, 0, stream>>>(score, mm, score2);
    colreduce_kernel<<<64, 256, 0, stream>>>(score2, mm, colmax, colsum, argmx);
    attzero_kernel <<<34848, 256, 0, stream>>>((unsigned int*)attp);
    attfill_kernel <<<dim3(16, 4096), 256, 0, stream>>>(score2, mm, colmax, colsum, attp);

    gemm2_kernel   <<<32, 256, 0, stream>>>(KT, attp, rec);

    final_kernel   <<<1024, 256, 0, stream>>>(rec, fg_b, m_b, out_b);
    offsets_kernel <<<16, 256, 0, stream>>>(argmx, out2_b);
  }
}
